// FP8LinearModule_78237124264320
// MI455X (gfx1250) — 8-bit matrix operand order repaired and checked by reading, compiled, not run on hardware
//
#include <hip/hip_runtime.h>
#include <hip/hip_bf16.h>
#include <stdint.h>

typedef __attribute__((ext_vector_type(16))) int   v16i;
typedef __attribute__((ext_vector_type(8)))  float v8f;

constexpr int Mdim = 8192;
constexpr int Kdim = 4096;
constexpr int Ndim = 4096;
constexpr int KBLK = Kdim / 128;
constexpr int TILE_BYTES = 16 * 128;

constexpr int A_STAGE = 8 * TILE_BYTES;
constexpr int B_STAGE = 128 * 128;
constexpr int STAGE   = A_STAGE + B_STAGE;
constexpr int LDS_TOTAL = 2 * STAGE;

__device__ __forceinline__ float bf16_lo(unsigned u) { return __uint_as_float(u << 16); }
__device__ __forceinline__ float bf16_hi(unsigned u) { return __uint_as_float(u & 0xFFFF0000u); }

#if __has_builtin(__builtin_amdgcn_cvt_pk_fp8_f32)
__device__ __forceinline__ unsigned pack4_fp8(float f0, float f1, float f2, float f3) {
    int p = 0;
    p = __builtin_amdgcn_cvt_pk_fp8_f32(f0, f1, p, false);
    p = __builtin_amdgcn_cvt_pk_fp8_f32(f2, f3, p, true);
    return (unsigned)p;
}
#else
__device__ __forceinline__ unsigned char f32_to_e4m3(float x) {
    unsigned u    = __float_as_uint(x);
    unsigned sign = (u >> 24) & 0x80u;
    float a = fabsf(x);
    if (a > 448.0f) return sign | 0x7E;
    unsigned bits = __float_as_uint(a);
    int e = (int)(bits >> 23) - 127;
    if (e < -10) return (unsigned char)sign;
    unsigned mant = bits & 0x7FFFFFu;
    unsigned q, rsh;
    if (e < -6) { mant |= 0x800000u; rsh = 20u + (unsigned)(-6 - e); q = mant >> rsh; }
    else        { rsh = 20u; q = ((unsigned)(e + 7) << 3) | (mant >> 20); }
    unsigned rem  = mant & ((1u << rsh) - 1u);
    unsigned half = 1u << (rsh - 1u);
    if (rem > half || (rem == half && (q & 1u))) q++;
    if (q > 0x7Eu) q = 0x7Eu;
    return (unsigned char)(sign | q);
}
__device__ __forceinline__ unsigned pack4_fp8(float f0, float f1, float f2, float f3) {
    return (unsigned)f32_to_e4m3(f0) | ((unsigned)f32_to_e4m3(f1) << 8) |
           ((unsigned)f32_to_e4m3(f2) << 16) | ((unsigned)f32_to_e4m3(f3) << 24);
}
#endif

#define ASYNC_CP16(lds_u32, gaddr_u64, OFFSTR)                                  \
    asm volatile("global_load_async_to_lds_b128 %0, %1, off offset:" OFFSTR     \
                 :: "v"(lds_u32), "v"(gaddr_u64) : "memory")

__device__ __forceinline__ unsigned lds_off_u32(const uint8_t* l) {
    return (unsigned)(uintptr_t)(__attribute__((address_space(3))) const uint8_t*)l;
}

__device__ __forceinline__ void copy64_g2l(const uint8_t* g, const uint8_t* l) {
    const unsigned           la = lds_off_u32(l);
    const unsigned long long ga = (unsigned long long)(uintptr_t)g;
    ASYNC_CP16(la, ga, "0");
    ASYNC_CP16(la, ga, "16");
    ASYNC_CP16(la, ga, "32");
    ASYNC_CP16(la, ga, "48");
}

__device__ __forceinline__ void wait_async_le8() {
#if __has_builtin(__builtin_amdgcn_s_wait_asynccnt)
    __builtin_amdgcn_s_wait_asynccnt(8);
#else
    asm volatile("s_wait_asynccnt 0x8" ::: "memory");
#endif
}
__device__ __forceinline__ void wait_async_le0() {
#if __has_builtin(__builtin_amdgcn_s_wait_asynccnt)
    __builtin_amdgcn_s_wait_asynccnt(0);
#else
    asm volatile("s_wait_asynccnt 0x0" ::: "memory");
#endif
}

__global__ __launch_bounds__(256) void quant_rows_kernel(
    const __hip_bfloat16* __restrict__ x,
    const __hip_bfloat16* __restrict__ w,
    uint8_t* __restrict__ xq, uint8_t* __restrict__ wq,
    float* __restrict__ sa, float* __restrict__ sw)
{
    const int wave = threadIdx.x >> 5;
    const int lane = threadIdx.x & 31;
    const int row  = blockIdx.x * 8 + wave;
    const bool is_x = row < Mdim;

    const uint8_t* srcb;
    float*         scp;
    if (is_x) { srcb = (const uint8_t*)(x + (size_t)row * Kdim); scp = sa + row; }
    else      { const int r = row - Mdim;
                srcb = (const uint8_t*)(w + (size_t)r * Kdim);   scp = sw + r; }

    const uint4* src4 = (const uint4*)srcb;
    uint4 vals[16];
    float amax = 0.0f;
#pragma unroll
    for (int c = 0; c < 16; ++c) {
        uint4 v = src4[c * 32 + lane];
        vals[c] = v;
        amax = fmaxf(amax, fabsf(bf16_lo(v.x))); amax = fmaxf(amax, fabsf(bf16_hi(v.x)));
        amax = fmaxf(amax, fabsf(bf16_lo(v.y))); amax = fmaxf(amax, fabsf(bf16_hi(v.y)));
        amax = fmaxf(amax, fabsf(bf16_lo(v.z))); amax = fmaxf(amax, fabsf(bf16_hi(v.z)));
        amax = fmaxf(amax, fabsf(bf16_lo(v.w))); amax = fmaxf(amax, fabsf(bf16_hi(v.w)));
    }
#pragma unroll
    for (int off = 16; off >= 1; off >>= 1)
        amax = fmaxf(amax, __shfl_xor(amax, off, 32));

    const float scale = fmaxf(amax * (1.0f / 448.0f), 1e-8f);
    const float rinv  = 1.0f / scale;
    if (lane == 0) *scp = scale;

    uint8_t* xtile = is_x ? (xq + (size_t)(row >> 4) * KBLK * TILE_BYTES) : nullptr;
    const int ml   = row & 15;
    uint2* wdst2   = is_x ? nullptr : (uint2*)(wq + (size_t)(row - Mdim) * Kdim);

#pragma unroll
    for (int c = 0; c < 16; ++c) {
        uint4 v = vals[c];
        uint2 q;
        q.x = pack4_fp8(bf16_lo(v.x) * rinv, bf16_hi(v.x) * rinv,
                        bf16_lo(v.y) * rinv, bf16_hi(v.y) * rinv);
        q.y = pack4_fp8(bf16_lo(v.z) * rinv, bf16_hi(v.z) * rinv,
                        bf16_lo(v.w) * rinv, bf16_hi(v.w) * rinv);
        const int g = c * 32 + lane;
        if (is_x) {
            const int kb = g >> 4;
            const int u  = g & 15;
            const size_t off = (size_t)kb * TILE_BYTES
                             + (size_t)((ml | (((u >> 1) & 1) << 4)) * 64)
                             + (size_t)((u >> 2) * 16 + (u & 1) * 8);
            *(uint2*)(xtile + off) = q;
        } else {
            wdst2[g] = q;
        }
    }
}

__device__ __forceinline__ v8f wmma_fp8(v16i a, v16i b, v8f c) {
    return __builtin_amdgcn_wmma_f32_16x16x128_fp8_fp8(a, b, (short)0, c, false, false);
}

__device__ __forceinline__ v16i load_frag_lds(const uint8_t* p, int stride) {
    v16i a;
#pragma unroll
    for (int i = 0; i < 4; ++i) {
        uint4 t = *(const uint4*)(p + i * stride);
        a[4 * i]     = (int)t.x;
        a[4 * i + 1] = (int)t.y;
        a[4 * i + 2] = (int)t.z;
        a[4 * i + 3] = (int)t.w;
    }
    return a;
}

__global__ __launch_bounds__(256) void fp8_gemm_kernel(
    const uint8_t* __restrict__ xq, const uint8_t* __restrict__ wq,
    const float* __restrict__ sa, const float* __restrict__ sw,
    const __hip_bfloat16* __restrict__ bias,
    __hip_bfloat16* __restrict__ out)
{
    extern __shared__ uint8_t lds_raw[];

    const int t    = threadIdx.x;
    const int lane = t & 31;
    const int wid  = t >> 5;
    const int wm   = wid & 1;
    const int wn   = wid >> 1;
    const int mb0    = blockIdx.y * 8;
    const int blockN = blockIdx.x * 128;
    const int tileM  = blockIdx.y * 128 + wm * 64;
    const int tileN  = blockN + wn * 32;

    const int  hl = lane & 15;
    const bool hi = lane >= 16;
    const int  kbB = hi ? 16 : 0;

    const int idx4 = 4 * t;
    const int fA   = idx4 >> 7;
    const int offA = (idx4 & 127) << 4;
    const int nB   = idx4 >> 3;
    const int offB = (idx4 & 7) << 4;

    const uint8_t* gA0 = xq + ((size_t)(mb0 + fA) * KBLK) * TILE_BYTES + offA;
    const uint8_t* gB0 = wq + (size_t)(blockN + nB) * Kdim + offB;
    const int      lA  = fA * TILE_BYTES + offA;
    const int      lB  = A_STAGE + nB * 128 + offB;

    v8f acc[4][2] = {};

    copy64_g2l(gA0,              lds_raw + lA);
    copy64_g2l(gB0,              lds_raw + lB);
    copy64_g2l(gA0 + TILE_BYTES, lds_raw + STAGE + lA);
    copy64_g2l(gB0 + 128,        lds_raw + STAGE + lB);

    for (int kb = 0; kb < KBLK; ++kb) {
        if (kb + 1 < KBLK) wait_async_le8();
        else               wait_async_le0();
        __syncthreads();

        const uint8_t* st = lds_raw + (kb & 1) * STAGE;
        v16i A[4], B[2];
#pragma unroll
        for (int f = 0; f < 4; ++f)
            A[f] = load_frag_lds(st + (wm * 4 + f) * TILE_BYTES + lane * 64, 16);
#pragma unroll
        for (int j = 0; j < 2; ++j)
            B[j] = load_frag_lds(st + A_STAGE + (wn * 32 + j * 16 + hl) * 128 + kbB, 32);

#pragma unroll
        for (int f = 0; f < 4; ++f) {
            acc[f][0] = wmma_fp8(A[f], B[0], acc[f][0]);
            acc[f][1] = wmma_fp8(A[f], B[1], acc[f][1]);
        }

        __syncthreads();
        if (kb + 2 < KBLK) {
            const size_t ks = (size_t)(kb + 2);
            copy64_g2l(gA0 + ks * TILE_BYTES, lds_raw + (kb & 1) * STAGE + lA);
            copy64_g2l(gB0 + ks * 128,        lds_raw + (kb & 1) * STAGE + lB);
        }
    }

    const int mofs = hi ? 8 : 0;
#pragma unroll
    for (int j = 0; j < 2; ++j) {
        const int   n   = tileN + j * 16 + hl;
        const float swv = sw[n];
        const float bv  = __bfloat162float(bias[n]);
#pragma unroll
        for (int f = 0; f < 4; ++f) {
            const v8f a = acc[f][j];
#pragma unroll
            for (int r = 0; r < 8; ++r) {
                const int m = tileM + f * 16 + r + mofs;
                const float o = a[r] * sa[m] * swv + bv;
                out[(size_t)m * Ndim + n] = __float2bfloat16(o);
            }
        }
    }
}

extern "C" void kernel_launch(void* const* d_in, const int* in_sizes, int n_in,
                              void* d_out, int out_size, void* d_ws, size_t ws_size,
                              hipStream_t stream) {
    (void)in_sizes; (void)n_in; (void)out_size; (void)ws_size;
    const __hip_bfloat16* x    = (const __hip_bfloat16*)d_in[0];
    const __hip_bfloat16* w    = (const __hip_bfloat16*)d_in[1];
    const __hip_bfloat16* bias = (const __hip_bfloat16*)d_in[2];

    uint8_t* ws = (uint8_t*)d_ws;
    uint8_t* xq = ws;
    uint8_t* wq = ws + (size_t)Mdim * Kdim;
    float*   sa = (float*)(wq + (size_t)Ndim * Kdim);
    float*   sw = sa + Mdim;

    quant_rows_kernel<<<(Mdim + Ndim) / 8, 256, 0, stream>>>(x, w, xq, wq, sa, sw);

    dim3 grid(Ndim / 128, Mdim / 128);
    fp8_gemm_kernel<<<grid, 256, LDS_TOTAL, stream>>>(xq, wq, sa, sw, bias,
                                                      (__hip_bfloat16*)d_out);
}
